// VQVAE_9208409882932
// MI455X (gfx1250) — matrix operand order repaired and checked by reading, not compiled, not run on hardware
//
#include <hip/hip_runtime.h>
#include <hip/hip_bf16.h>
#include <stddef.h>

// ---------------------------------------------------------------------------
// VQ-VAE (BiLSTM encoder/decoder + vector quantization) for gfx1250 / MI455X.
// Time-parallel GEMMs -> v_wmma_f32_16x16x32_f16 (f32 accum), double-buffered
// LDS pipeline. Sequential LSTM recurrence -> persistent VALU GEMV kernel.
// ---------------------------------------------------------------------------

typedef __attribute__((ext_vector_type(16))) _Float16 v16h;
typedef __attribute__((ext_vector_type(8)))  _Float16 v8h;
typedef __attribute__((ext_vector_type(2)))  _Float16 h2;
typedef __attribute__((ext_vector_type(8)))  float    v8f;

#define T_FRM 4096
#define M_MORA 512
#define K_CODE 512
#define ZD 64
#define HS 400
#define G4H 1600
#define LING 442
#define AC 199
#define ENC_IN 641
#define DEC_IN 506
#define NH 800

// ------------------------- WMMA GEMM:  C = act(A @ W^T + b1 + b2) ----------
// A: [M,K] f32 row-major (M must be a multiple of 128 -- true for all calls).
// W: [N,K] f32 row-major (PyTorch weight layout).
// 256 threads = 8 waves; block tile 128x128; K tile 32; f32->f16 staged
// through double-buffered LDS; one barrier per K chunk.
#define TM 128
#define TN 128
#define TK 32
#define LPITCH 40   // halves; 80B row pitch, 16B aligned, bank-spread

// fetch 128x32 A tile into 16 regs (2 consecutive cols per thread, coalesced)
__device__ __forceinline__ void fetch_tileA(const float* __restrict__ A,
        int rowBase, int k0, int K, int tid, float* v)
{
    const int c  = (tid * 2) & 31;
    const int rb = tid >> 4;
    if (k0 + TK <= K) {                       // interior chunk: no guards
#pragma unroll
        for (int j = 0; j < 8; ++j) {
            const size_t base = (size_t)(rowBase + rb + 16 * j) * K + k0 + c;
            v[2 * j]     = A[base];
            v[2 * j + 1] = A[base + 1];
        }
    } else {                                  // K edge: clamp addr, select 0
#pragma unroll
        for (int j = 0; j < 8; ++j) {
            const size_t rB = (size_t)(rowBase + rb + 16 * j) * K;
            const int gc0 = k0 + c, gc1 = gc0 + 1;
            const int c0 = gc0 < K ? gc0 : K - 1;
            const int c1 = gc1 < K ? gc1 : K - 1;
            float t0 = A[rB + c0];
            float t1 = A[rB + c1];
            v[2 * j]     = gc0 < K ? t0 : 0.f;
            v[2 * j + 1] = gc1 < K ? t1 : 0.f;
        }
    }
}

// fetch 128x32 W tile (rows are output cols; needs N-edge guard too)
__device__ __forceinline__ void fetch_tileB(const float* __restrict__ W,
        int colBase, int k0, int N, int K, int tid, float* v)
{
    const int c  = (tid * 2) & 31;
    const int rb = tid >> 4;
    if ((k0 + TK <= K) && (colBase + TN <= N)) {
#pragma unroll
        for (int j = 0; j < 8; ++j) {
            const size_t base = (size_t)(colBase + rb + 16 * j) * K + k0 + c;
            v[2 * j]     = W[base];
            v[2 * j + 1] = W[base + 1];
        }
    } else {
#pragma unroll
        for (int j = 0; j < 8; ++j) {
            const int gr = colBase + rb + 16 * j;
            const int rc = gr < N ? gr : N - 1;
            const int gc0 = k0 + c, gc1 = gc0 + 1;
            const int c0 = gc0 < K ? gc0 : K - 1;
            const int c1 = gc1 < K ? gc1 : K - 1;
            float t0 = W[(size_t)rc * K + c0];
            float t1 = W[(size_t)rc * K + c1];
            v[2 * j]     = (gr < N && gc0 < K) ? t0 : 0.f;
            v[2 * j + 1] = (gr < N && gc1 < K) ? t1 : 0.f;
        }
    }
}

// convert 16 staged floats -> packed f16 pairs in LDS (b32 stores)
__device__ __forceinline__ void store_tile(_Float16* S, int tid, const float* v)
{
    const int c  = (tid * 2) & 31;
    const int rb = tid >> 4;
#pragma unroll
    for (int j = 0; j < 8; ++j) {
        h2 p;
        p[0] = (_Float16)v[2 * j];
        p[1] = (_Float16)v[2 * j + 1];
        *(h2*)(S + (rb + 16 * j) * LPITCH + c) = p;
    }
}

__device__ __forceinline__ void compute_chunk(const _Float16* As, const _Float16* Bs,
        int wm, int wn, int lane16, int g, v8f acc[4][2])
{
    // B fragments: lane = N column, halves sweep K (K = slot + 16*g)
    v16h bfrag[2];
#pragma unroll
    for (int nt = 0; nt < 2; ++nt) {
        const _Float16* pb = &Bs[(wn * 32 + nt * 16 + lane16) * LPITCH + g * 16];
        v8h blo = *(const v8h*)pb;
        v8h bhi = *(const v8h*)(pb + 8);
#pragma unroll
        for (int e = 0; e < 8; ++e) { bfrag[nt][e] = blo[e]; bfrag[nt][8 + e] = bhi[e]; }
    }
#pragma unroll
    for (int mt = 0; mt < 4; ++mt) {
        // A fragment: M = lane16; K = g*16 + h
        const _Float16* pa = &As[(wm * 64 + mt * 16 + lane16) * LPITCH + g * 16];
        v8h alo = *(const v8h*)pa;
        v8h ahi = *(const v8h*)(pa + 8);
        v16h afrag;
#pragma unroll
        for (int e = 0; e < 8; ++e) { afrag[e] = alo[e]; afrag[8 + e] = ahi[e]; }
#pragma unroll
        for (int nt = 0; nt < 2; ++nt) {
            acc[mt][nt] = __builtin_amdgcn_wmma_f32_16x16x32_f16(
                false, afrag, false, bfrag[nt], (short)0, acc[mt][nt], false, false);
        }
    }
}

__global__ __launch_bounds__(256) void wmma_gemm_kernel(
    const float* __restrict__ A, const float* __restrict__ W,
    const float* __restrict__ b1, const float* __restrict__ b2,
    float* __restrict__ C, int M, int N, int K, int relu)
{
    __shared__ __align__(16) _Float16 As[2][TM * LPITCH];
    __shared__ __align__(16) _Float16 Bs[2][TN * LPITCH];

    const int tid    = threadIdx.x;
    const int lane   = tid & 31;
    const int wave   = tid >> 5;
    const int lane16 = lane & 15;
    const int g      = lane >> 4;          // half-wave id
    const int wm     = wave >> 2;          // 0..1 : 64-row strip
    const int wn     = wave & 3;           // 0..3 : 32-col strip
    const int rowBase = blockIdx.y * TM;
    const int colBase = blockIdx.x * TN;

    const v8f zf = {0.f,0.f,0.f,0.f,0.f,0.f,0.f,0.f};
    v8f acc[4][2];
#pragma unroll
    for (int i = 0; i < 4; ++i)
#pragma unroll
        for (int j = 0; j < 2; ++j) acc[i][j] = zf;

    float av[16], bv[16];
    const int kTiles = (K + TK - 1) / TK;

    // prologue: stage chunk 0
    fetch_tileA(A, rowBase, 0, K, tid, av);
    fetch_tileB(W, colBase, 0, N, K, tid, bv);
    store_tile(As[0], tid, av);
    store_tile(Bs[0], tid, bv);
    __syncthreads();

    for (int kt = 0; kt < kTiles; ++kt) {
        const int cur = kt & 1;
        const bool hasNext = (kt + 1) < kTiles;
        if (hasNext) {                       // issue next chunk's global loads
            fetch_tileA(A, rowBase, (kt + 1) * TK, K, tid, av);
            fetch_tileB(W, colBase, (kt + 1) * TK, N, K, tid, bv);
        }
        compute_chunk(As[cur], Bs[cur], wm, wn, lane16, g, acc);  // overlap
        if (hasNext) {                       // convert + park in other buffer
            store_tile(As[cur ^ 1], tid, av);
            store_tile(Bs[cur ^ 1], tid, bv);
        }
        __syncthreads();
    }

    // store: C/D layout  M = r + 8*g (VGPR r), N = lane16
#pragma unroll
    for (int mt = 0; mt < 4; ++mt) {
#pragma unroll
        for (int nt = 0; nt < 2; ++nt) {
            int col = colBase + wn * 32 + nt * 16 + lane16;
            if (col < N) {
                float bias = 0.f;
                if (b1) bias += b1[col];
                if (b2) bias += b2[col];
#pragma unroll
                for (int r = 0; r < 8; ++r) {
                    int row = rowBase + wm * 64 + mt * 16 + g * 8 + r;
                    if (row < M) {
                        float v = acc[mt][nt][r] + bias;
                        if (relu) v = fmaxf(v, 0.f);
                        C[(size_t)row * N + col] = v;
                    }
                }
            }
        }
    }
}

// ------------------------- LSTM recurrence (persistent, per-direction) -----
__device__ __forceinline__ float sigf(float x) {
    return 1.f / (1.f + __expf(-x));
}
__device__ __forceinline__ float tanhfast(float x) {
    x = fminf(fmaxf(x, -15.f), 15.f);
    float e = __expf(2.f * x);
    return (e - 1.f) / (e + 1.f);
}

// grid = 2 (dir), block = 1024.  xg holds x@Wih^T + bih + bhh for every t.
// whhT: [dir][k=0..399][r=0..1599]  (pre-transposed for coalescing).
// hout: [T, 800]; dir 0 writes cols 0..399, dir 1 cols 400..799.
__global__ __launch_bounds__(1024) void lstm_recur_kernel(
    const float* __restrict__ xg_f, const float* __restrict__ xg_b,
    const float* __restrict__ whhT, float* __restrict__ hout, int relu_store)
{
    __shared__ float h_s[HS];
    __shared__ float g_s[G4H];
    const int tid = threadIdx.x;
    const int dir = blockIdx.x;
    const float* xg = dir ? xg_b : xg_f;
    const float* wT = whhT + (size_t)dir * HS * G4H;

    if (tid < HS) h_s[tid] = 0.f;
    float c = 0.f;
    __syncthreads();

    const int r0 = tid;
    const int r1 = tid + 1024;
    const bool two = (r1 < G4H);

    for (int s = 0; s < T_FRM; ++s) {
        const int t = dir ? (T_FRM - 1 - s) : s;
        const float* xgt = xg + (size_t)t * G4H;
        float acc0 = xgt[r0];
        float acc1 = two ? xgt[r1] : 0.f;
        if (two) {
#pragma unroll 4
            for (int k = 0; k < HS; ++k) {
                float hk = h_s[k];
                const float* wk = wT + (size_t)k * G4H;
                acc0 = fmaf(wk[r0], hk, acc0);
                acc1 = fmaf(wk[r1], hk, acc1);
            }
        } else {
#pragma unroll 4
            for (int k = 0; k < HS; ++k)
                acc0 = fmaf(wT[(size_t)k * G4H + r0], h_s[k], acc0);
        }
        g_s[r0] = acc0;
        if (two) g_s[r1] = acc1;
        __syncthreads();
        if (tid < HS) {
            float ig = sigf(g_s[tid]);
            float fg = sigf(g_s[HS + tid]);
            float gg = tanhfast(g_s[2 * HS + tid]);
            float og = sigf(g_s[3 * HS + tid]);
            c = fg * c + ig * gg;
            float h = og * tanhfast(c);
            h_s[tid] = h;
            hout[(size_t)t * NH + dir * HS + tid] = relu_store ? fmaxf(h, 0.f) : h;
        }
        __syncthreads();
    }
}

// ------------------------- glue kernels -------------------------------------
__global__ void pack_cat_kernel(const float* __restrict__ ling,
                                const float* __restrict__ ac,
                                float* __restrict__ out, int n)
{
    int i = blockIdx.x * blockDim.x + threadIdx.x;
    if (i >= n) return;
    int row = i / ENC_IN, col = i % ENC_IN;
    out[i] = (col < LING) ? ling[(size_t)row * LING + col]
                          : ac[(size_t)row * AC + (col - LING)];
}

__global__ void transpose_whh_kernel(const float* __restrict__ whh,
                                     float* __restrict__ whhT, int n)
{
    int i = blockIdx.x * blockDim.x + threadIdx.x;
    if (i >= n) return;                       // n = 2*1600*400
    int dir = i / (G4H * HS);
    int rem = i - dir * (G4H * HS);
    int r = rem / HS, k = rem % HS;
    whhT[(size_t)dir * HS * G4H + (size_t)k * G4H + r] = whh[i];
}

__global__ void gather_relu_kernel(const float* __restrict__ h,
                                   const int* __restrict__ mora,
                                   float* __restrict__ h1)
{
    int m = blockIdx.x;
    int row = mora[m];
    for (int j = threadIdx.x; j < NH; j += blockDim.x)
        h1[(size_t)m * NH + j] = fmaxf(h[(size_t)row * NH + j], 0.f);
}

__global__ void cnorm_kernel(const float* __restrict__ cb, float* __restrict__ cn)
{
    int i = blockIdx.x * blockDim.x + threadIdx.x;
    if (i >= K_CODE) return;
    float s = 0.f;
    for (int j = 0; j < ZD; ++j) { float v = cb[(size_t)i * ZD + j]; s = fmaf(v, v, s); }
    cn[i] = s;
}

// argmin_k ( ||c_k||^2 - 2 * <znq_m, c_k> ) then select codebook row -> z
__global__ __launch_bounds__(256) void argmin_select_kernel(
    const float* __restrict__ scores, const float* __restrict__ cn,
    const float* __restrict__ cb, float* __restrict__ z)
{
    __shared__ float vals[256];
    __shared__ int   idxs[256];
    const int m = blockIdx.x, tid = threadIdx.x;
    float best = 3.4e38f; int bi = 0;
    for (int k = tid; k < K_CODE; k += 256) {
        float v = cn[k] - 2.f * scores[(size_t)m * K_CODE + k];
        if (v < best) { best = v; bi = k; }
    }
    vals[tid] = best; idxs[tid] = bi;
    __syncthreads();
    for (int s = 128; s > 0; s >>= 1) {
        if (tid < s) {
            if (vals[tid + s] < vals[tid] ||
                (vals[tid + s] == vals[tid] && idxs[tid + s] < idxs[tid])) {
                vals[tid] = vals[tid + s]; idxs[tid] = idxs[tid + s];
            }
        }
        __syncthreads();
    }
    const int k = idxs[0];
    if (tid < ZD) z[(size_t)m * ZD + tid] = cb[(size_t)k * ZD + tid];
}

// x_dec[t] = [ ling[t] | z[seg(t)] ] with seg = searchsorted(mora, t, right)
__global__ void expand_pack_kernel(const float* __restrict__ ling,
                                   const float* __restrict__ z,
                                   const int* __restrict__ mora,
                                   float* __restrict__ xdec)
{
    __shared__ int seg_s;
    const int t = blockIdx.x;
    if (threadIdx.x == 0) {
        int lo = 0, hi = M_MORA;
        while (lo < hi) { int mid = (lo + hi) >> 1; if (mora[mid] <= t) lo = mid + 1; else hi = mid; }
        seg_s = lo;
    }
    __syncthreads();
    const int seg = seg_s;
    for (int j = threadIdx.x; j < DEC_IN; j += blockDim.x) {
        float v;
        if (j < LING) v = ling[(size_t)t * LING + j];
        else          v = (seg < M_MORA) ? z[(size_t)seg * ZD + (j - LING)] : 0.f;
        xdec[(size_t)t * DEC_IN + j] = v;
    }
}

// ---------------------------------------------------------------------------
static inline void gemm(const float* A, const float* W, const float* b1,
                        const float* b2, float* C, int M, int N, int K,
                        int relu, hipStream_t s)
{
    dim3 g((N + TN - 1) / TN, (M + TM - 1) / TM);
    wmma_gemm_kernel<<<g, 256, 0, s>>>(A, W, b1, b2, C, M, N, K, relu);
}

extern "C" void kernel_launch(void* const* d_in, const int* in_sizes, int n_in,
                              void* d_out, int out_size, void* d_ws, size_t ws_size,
                              hipStream_t stream)
{
    const float* ling     = (const float*)d_in[0];
    const float* ac       = (const float*)d_in[1];
    const int*   mora     = (const int*)  d_in[2];
    const float* fc11_w   = (const float*)d_in[3];
    const float* fc11_b   = (const float*)d_in[4];
    const float* fc2_w    = (const float*)d_in[5];
    const float* fc2_b    = (const float*)d_in[6];
    const float* fc12_w   = (const float*)d_in[7];
    const float* fc12_b   = (const float*)d_in[8];
    const float* fc3_w    = (const float*)d_in[9];
    const float* fc3_b    = (const float*)d_in[10];
    const float* codebook = (const float*)d_in[11];
    const float* l1_Wih0  = (const float*)d_in[12];
    const float* l1_Whh0  = (const float*)d_in[13];
    const float* l1_bih0  = (const float*)d_in[14];
    const float* l1_bhh0  = (const float*)d_in[15];
    const float* l1_Wih1  = (const float*)d_in[16];
    const float* l1_Whh1  = (const float*)d_in[17];
    const float* l1_bih1  = (const float*)d_in[18];
    const float* l1_bhh1  = (const float*)d_in[19];
    const float* l2_Wih0  = (const float*)d_in[20];
    const float* l2_Whh0  = (const float*)d_in[21];
    const float* l2_bih0  = (const float*)d_in[22];
    const float* l2_bhh0  = (const float*)d_in[23];
    const float* l2_Wih1  = (const float*)d_in[24];
    const float* l2_Whh1  = (const float*)d_in[25];
    const float* l2_bih1  = (const float*)d_in[26];
    const float* l2_bhh1  = (const float*)d_in[27];

    // output regions: dec [4096,199] | z [512,64] | znq [512,64]
    float* dec = (float*)d_out;
    float* zq  = dec + (size_t)T_FRM * AC;
    float* znq = zq + (size_t)M_MORA * ZD;

    // workspace layout (floats); buf0/buf1 reused between enc and dec phases
    float* ws = (float*)d_ws;
    size_t o = 0;
    float* buf0  = ws + o; o += (size_t)T_FRM * ENC_IN;  // x_cat -> xd_enc
    float* buf1  = ws + o; o += (size_t)T_FRM * ENC_IN;  // x_enc -> x_dec
    float* xg_f  = ws + o; o += (size_t)T_FRM * G4H;
    float* xg_b  = ws + o; o += (size_t)T_FRM * G4H;
    float* h_a   = ws + o; o += (size_t)T_FRM * NH;
    float* h_b   = ws + o; o += (size_t)T_FRM * NH;
    float* whhT  = ws + o; o += (size_t)2 * HS * G4H;
    float* h1    = ws + o; o += (size_t)M_MORA * NH;
    float* scr   = ws + o; o += (size_t)M_MORA * K_CODE;
    float* cn    = ws + o; o += (size_t)K_CODE;
    (void)ws_size; (void)n_in; (void)in_sizes; (void)out_size;

    const int nWhh = 2 * G4H * HS;

    // ---- encoder ----
    {
        int n = T_FRM * ENC_IN;
        pack_cat_kernel<<<(n + 255) / 256, 256, 0, stream>>>(ling, ac, buf0, n);
    }
    gemm(buf0, fc11_w, fc11_b, nullptr, buf1, T_FRM, ENC_IN, ENC_IN, 1, stream);

    // lstm1 layer 0 (in 641)
    transpose_whh_kernel<<<(nWhh + 255) / 256, 256, 0, stream>>>(l1_Whh0, whhT, nWhh);
    gemm(buf1, l1_Wih0,                      l1_bih0,       l1_bhh0,       xg_f, T_FRM, G4H, ENC_IN, 0, stream);
    gemm(buf1, l1_Wih0 + (size_t)G4H*ENC_IN, l1_bih0 + G4H, l1_bhh0 + G4H, xg_b, T_FRM, G4H, ENC_IN, 0, stream);
    lstm_recur_kernel<<<2, 1024, 0, stream>>>(xg_f, xg_b, whhT, h_a, 0);

    // lstm1 layer 1 (in 800)
    transpose_whh_kernel<<<(nWhh + 255) / 256, 256, 0, stream>>>(l1_Whh1, whhT, nWhh);
    gemm(h_a, l1_Wih1,                  l1_bih1,       l1_bhh1,       xg_f, T_FRM, G4H, NH, 0, stream);
    gemm(h_a, l1_Wih1 + (size_t)G4H*NH, l1_bih1 + G4H, l1_bhh1 + G4H, xg_b, T_FRM, G4H, NH, 0, stream);
    lstm_recur_kernel<<<2, 1024, 0, stream>>>(xg_f, xg_b, whhT, h_b, 0);

    // mora gather + fc2 -> znq (written straight into d_out region)
    gather_relu_kernel<<<M_MORA, 256, 0, stream>>>(h_b, mora, h1);
    gemm(h1, fc2_w, fc2_b, nullptr, znq, M_MORA, ZD, NH, 0, stream);

    // ---- vector quantization ----
    cnorm_kernel<<<(K_CODE + 255) / 256, 256, 0, stream>>>(codebook, cn);
    gemm(znq, codebook, nullptr, nullptr, scr, M_MORA, K_CODE, ZD, 0, stream);
    argmin_select_kernel<<<M_MORA, 256, 0, stream>>>(scr, cn, codebook, zq);

    // ---- decoder ----
    expand_pack_kernel<<<T_FRM, 128, 0, stream>>>(ling, zq, mora, buf1);      // x_dec
    gemm(buf1, fc12_w, fc12_b, nullptr, buf0, T_FRM, DEC_IN, DEC_IN, 1, stream); // xd_enc

    // lstm2 layer 0 (in 506)
    transpose_whh_kernel<<<(nWhh + 255) / 256, 256, 0, stream>>>(l2_Whh0, whhT, nWhh);
    gemm(buf0, l2_Wih0,                      l2_bih0,       l2_bhh0,       xg_f, T_FRM, G4H, DEC_IN, 0, stream);
    gemm(buf0, l2_Wih0 + (size_t)G4H*DEC_IN, l2_bih0 + G4H, l2_bhh0 + G4H, xg_b, T_FRM, G4H, DEC_IN, 0, stream);
    lstm_recur_kernel<<<2, 1024, 0, stream>>>(xg_f, xg_b, whhT, h_a, 0);

    // lstm2 layer 1 (in 800), stores relu(h) for fc3
    transpose_whh_kernel<<<(nWhh + 255) / 256, 256, 0, stream>>>(l2_Whh1, whhT, nWhh);
    gemm(h_a, l2_Wih1,                  l2_bih1,       l2_bhh1,       xg_f, T_FRM, G4H, NH, 0, stream);
    gemm(h_a, l2_Wih1 + (size_t)G4H*NH, l2_bih1 + G4H, l2_bhh1 + G4H, xg_b, T_FRM, G4H, NH, 0, stream);
    lstm_recur_kernel<<<2, 1024, 0, stream>>>(xg_f, xg_b, whhT, h_b, 1);

    // fc3 -> dec output
    gemm(h_b, fc3_w, fc3_b, nullptr, dec, T_FRM, AC, NH, 0, stream);
}
